// CausalSelfAttention_54202487275835
// MI455X (gfx1250) — matrix operand order repaired and checked by reading, not compiled, not run on hardware
//
#include <hip/hip_runtime.h>
#include <hip/hip_bf16.h>
#include <stdint.h>

typedef __attribute__((ext_vector_type(16))) _Float16 v16h;
typedef __attribute__((ext_vector_type(8)))  _Float16 v8h;
typedef __attribute__((ext_vector_type(8)))  float    v8f;

#define CAT16(lo, hi) __builtin_shufflevector((lo), (hi), 0,1,2,3,4,5,6,7,8,9,10,11,12,13,14,15)

#define BB 2
#define TT 2048
#define CC 1024
#define HH 16
#define DD 64
#define MROWS (BB * TT)   // 4096
#define NKQV (3 * CC)     // 3072

static __device__ __forceinline__ v8f wmma_f16(v16h a, v16h b, v8f c) {
  // D = A(16x32 f16) * B(32x16 f16) + C(16x16 f32)
  return __builtin_amdgcn_wmma_f32_16x16x32_f16(false, a, false, b, (short)0, c, false, false);
}

// Async copy of one 16-byte chunk: global -> LDS, tracked by ASYNCcnt.
static __device__ __forceinline__ void async_copy_b128(uint32_t lds_byte_off, const void* gptr) {
  uint64_t ga = (uint64_t)(uintptr_t)gptr;
  asm volatile("global_load_async_to_lds_b128 %0, %1, off"
               :: "v"(lds_byte_off), "v"(ga) : "memory");
}

// ---------------- conversion kernels ----------------

__global__ __launch_bounds__(256) void f32_to_f16_kernel(const float* __restrict__ in,
                                                         _Float16* __restrict__ out, int n) {
  int i = blockIdx.x * 256 + threadIdx.x;
  if (i < n) out[i] = (_Float16)in[i];
}

// in[rows][cols] (f32, row-major)  ->  out[cols][rows] (f16)
__global__ __launch_bounds__(256) void transpose_f16_kernel(const float* __restrict__ in,
                                                            _Float16* __restrict__ out,
                                                            int rows, int cols) {
  int i = blockIdx.x * 256 + threadIdx.x;
  if (i < rows * cols) {
    int r = i / cols, c = i % cols;
    out[(size_t)c * rows + r] = (_Float16)in[i];
  }
}

// ---------------- QKV GEMM: [4096 x 1024] @ [1024 x 3072] + bias ----------------

__global__ __launch_bounds__(128) void qkv_gemm_kernel(
    const _Float16* __restrict__ xh, const _Float16* __restrict__ wt,
    const float* __restrict__ bias,
    _Float16* __restrict__ qh, float* __restrict__ kout, _Float16* __restrict__ kh,
    float* __restrict__ vout, _Float16* __restrict__ vth) {
  const int lane   = threadIdx.x & 31;
  const int wid    = threadIdx.x >> 5;
  const int tile   = blockIdx.x * 4 + wid;        // 0..12287
  const int nTile  = tile % (NKQV / 64);          // 48 tiles of 64 cols
  const int mTile  = tile / (NKQV / 64);
  const int row0   = mTile * 16;
  const int col0   = nTile * 64;
  const int laneHi = lane >> 4;
  const int ln     = lane & 15;

  v8f acc[4] = {};
  const _Float16* arow = xh + (size_t)(row0 + ln) * CC;
  for (int k0 = 0; k0 < CC; k0 += 32) {
    const int ab = k0 + (laneHi ? 16 : 0);
    v8h alo = *(const v8h*)(arow + ab);
    v8h ahi = *(const v8h*)(arow + ab + 8);
    v16h afrag = CAT16(alo, ahi);
#pragma unroll
    for (int t = 0; t < 4; ++t) {
      const _Float16* bp = wt + (size_t)(col0 + t * 16 + ln) * CC + k0 + (laneHi ? 16 : 0);
      v16h bfrag = *(const v16h*)bp;
      acc[t] = wmma_f16(afrag, bfrag, acc[t]);
    }
  }
#pragma unroll
  for (int t = 0; t < 4; ++t) {
    const int j     = col0 + t * 16 + ln;   // global column, same for all 8 rows in lane
    const float bj  = bias[j];
    const int which = j >> 10;              // 0=q, 1=k, 2=v
    const int jj    = j & (CC - 1);
    const int h     = jj >> 6;
    const int d     = jj & (DD - 1);
#pragma unroll
    for (int r = 0; r < 8; ++r) {
      const int row = row0 + r + laneHi * 8;
      const int b   = row >> 11;            // row / TT
      const int t_  = row & (TT - 1);
      const float val = acc[t][r] + bj;
      const size_t bhtd = (((size_t)(b * HH + h)) * TT + t_) * DD + d;
      if (which == 0) {
        qh[bhtd] = (_Float16)val;
      } else if (which == 1) {
        kout[bhtd] = val;
        kh[bhtd]   = (_Float16)val;
      } else {
        vout[bhtd] = val;
        vth[(((size_t)(b * HH + h)) * DD + d) * TT + t_] = (_Float16)val;
      }
    }
  }
}

// ---------------- Flash attention ----------------
// One workgroup (4 waves) = one (b, h) and 4 consecutive 16-query blocks.
// K/V tiles for each 32-key block are staged cooperatively in LDS with
// global_load_async_to_lds_b128 (double-buffered), shared by all 4 waves.

__global__ __launch_bounds__(128) void attn_kernel(
    const _Float16* __restrict__ qh, const _Float16* __restrict__ kh,
    const _Float16* __restrict__ vth, _Float16* __restrict__ yh) {
  __shared__ __align__(32) _Float16 Kt[2][32 * 64];   // [key][d]      4KB each
  __shared__ __align__(32) _Float16 Vt[2][64 * 32];   // [d][key]      4KB each
  __shared__ __align__(32) _Float16 pbuf_all[4 * 16 * 32];

  const int tid  = threadIdx.x;
  const int lane = tid & 31;
  const int wid  = tid >> 5;
  _Float16* pbuf = pbuf_all + wid * 512;

  const int grp = blockIdx.x;               // 0..1023
  const int qg  = grp & 31;                 // 32 groups of 4 q-blocks
  const int h   = (grp >> 5) & (HH - 1);
  const int b   = grp >> 9;
  const int qb  = qg * 4 + wid;
  const int q0  = qb * 16;

  const int laneHi = lane >> 4;
  const int ln     = lane & 15;

  const size_t bh = (size_t)(b * HH + h);
  const _Float16* qbase = qh  + bh * (size_t)TT * DD;
  const _Float16* kbase = kh  + bh * (size_t)TT * DD;
  const _Float16* vbase = vth + bh * (size_t)DD * TT;

  const uint32_t kt_lds[2] = { (uint32_t)(uintptr_t)&Kt[0][0], (uint32_t)(uintptr_t)&Kt[1][0] };
  const uint32_t vt_lds[2] = { (uint32_t)(uintptr_t)&Vt[0][0], (uint32_t)(uintptr_t)&Vt[1][0] };

  // Q fragments: row m = ln, head-dim as two 32-wide K chunks
  v16h qfrag[2];
  {
    const _Float16* qrow = qbase + (size_t)(q0 + ln) * DD;
#pragma unroll
    for (int c = 0; c < 2; ++c) {
      const int base = c * 32 + (laneHi ? 16 : 0);
      v8h lo = *(const v8h*)(qrow + base);
      v8h hi = *(const v8h*)(qrow + base + 8);
      qfrag[c] = CAT16(lo, hi);
    }
  }

  v8f acc[4] = {};
  float mstat[8], lstat[8];
#pragma unroll
  for (int r = 0; r < 8; ++r) { mstat[r] = -1e30f; lstat[r] = 0.0f; }

  const int myKblocks  = (q0 + 15) / 32 + 1;
  const int maxKblocks = ((qg * 4 + 3) * 16 + 15) / 32 + 1;

  // --- stage key-block kb into LDS buffer `bufi` (all 128 threads) ---
  auto stage = [&](int kb, int bufi) {
    const int k0 = kb * 32;
    // K tile: 32 keys x 64 d = 4096B, contiguous in kh
    const _Float16* ks = kbase + (size_t)k0 * DD;
#pragma unroll
    for (int c = tid; c < 256; c += 128)
      async_copy_b128(kt_lds[bufi] + c * 16, ks + c * 8);
    // V tile: 64 d-rows x 32 keys; each row segment = 64B = 4 chunks
#pragma unroll
    for (int c = tid; c < 256; c += 128) {
      const int d = c >> 2;
      async_copy_b128(vt_lds[bufi] + c * 16, vbase + (size_t)d * TT + k0 + (c & 3) * 8);
    }
  };

  stage(0, 0);
  for (int kb = 0; kb < maxKblocks; ++kb) {
    const int bufi = kb & 1;
    asm volatile("s_wait_asynccnt 0x0" ::: "memory");   // my async loads into bufi done
    __syncthreads();                                    // everyone's loads visible
    if (kb + 1 < maxKblocks) stage(kb + 1, bufi ^ 1);   // prefetch next tile

    if (kb < myKblocks) {
      const int k0 = kb * 32;
      const _Float16* ktile = &Kt[bufi][0];
      const _Float16* vtile = &Vt[bufi][0];

      // S = (Q K^T), two 16x16 key tiles, K-dim = head dim (2 chunks)
      v8f s[2] = {};
#pragma unroll
      for (int t = 0; t < 2; ++t) {
#pragma unroll
        for (int c = 0; c < 2; ++c) {
          v16h bf = *(const v16h*)(ktile + (t * 16 + ln) * DD + c * 32 + (laneHi ? 16 : 0));
          s[t] = wmma_f16(qfrag[c], bf, s[t]);
        }
      }
      // scale + causal mask
#pragma unroll
      for (int t = 0; t < 2; ++t) {
        const int kj = k0 + t * 16 + ln;
#pragma unroll
        for (int r = 0; r < 8; ++r) {
          const int qi = q0 + r + laneHi * 8;
          const float v = s[t][r] * 0.125f;   // 1/sqrt(64)
          s[t][r] = (kj > qi) ? -1e30f : v;
        }
      }
      // online softmax update
#pragma unroll
      for (int r = 0; r < 8; ++r) {
        float mv = fmaxf(s[0][r], s[1][r]);
        mv = fmaxf(mv, __shfl_xor(mv, 1, 32));
        mv = fmaxf(mv, __shfl_xor(mv, 2, 32));
        mv = fmaxf(mv, __shfl_xor(mv, 4, 32));
        mv = fmaxf(mv, __shfl_xor(mv, 8, 32));
        const float mnew = fmaxf(mstat[r], mv);
        const float p0 = __expf(s[0][r] - mnew);
        const float p1 = __expf(s[1][r] - mnew);
        s[0][r] = p0; s[1][r] = p1;
        float sum = p0 + p1;
        sum += __shfl_xor(sum, 1, 32);
        sum += __shfl_xor(sum, 2, 32);
        sum += __shfl_xor(sum, 4, 32);
        sum += __shfl_xor(sum, 8, 32);
        const float corr = __expf(mstat[r] - mnew);
        lstat[r] = lstat[r] * corr + sum;
        mstat[r] = mnew;
#pragma unroll
        for (int dt = 0; dt < 4; ++dt) acc[dt][r] *= corr;
      }
      // repack P: C-layout -> per-wave LDS slice -> A-layout fragment
#pragma unroll
      for (int t = 0; t < 2; ++t)
#pragma unroll
        for (int r = 0; r < 8; ++r)
          pbuf[(r + laneHi * 8) * 32 + t * 16 + ln] = (_Float16)s[t][r];
      asm volatile("s_wait_dscnt 0x0" ::: "memory");
      v16h pfrag;
      {
        const int pb = laneHi ? 16 : 0;
        v8h lo = *(const v8h*)(pbuf + ln * 32 + pb);
        v8h hi = *(const v8h*)(pbuf + ln * 32 + 8 + pb);
        pfrag = CAT16(lo, hi);
      }
      // acc += P @ V
#pragma unroll
      for (int dt = 0; dt < 4; ++dt) {
        v16h vf = *(const v16h*)(vtile + (dt * 16 + ln) * 32 + (laneHi ? 16 : 0));
        acc[dt] = wmma_f16(pfrag, vf, acc[dt]);
      }
    }
    __syncthreads();   // all reads of bufi done before it is overwritten
  }

  // finalize: y = acc / l, stored f16 in [B, T, C] for the projection GEMM
#pragma unroll
  for (int dt = 0; dt < 4; ++dt) {
#pragma unroll
    for (int r = 0; r < 8; ++r) {
      const int t_ = q0 + r + laneHi * 8;
      const float y = acc[dt][r] / lstat[r];
      yh[((size_t)b * TT + t_) * CC + h * DD + dt * 16 + ln] = (_Float16)y;
    }
  }
}

// ---------------- Projection GEMM: [4096 x 1024] @ [1024 x 1024] + bias -> f32 out ----------------

__global__ __launch_bounds__(128) void proj_gemm_kernel(
    const _Float16* __restrict__ yh, const _Float16* __restrict__ wpt,
    const float* __restrict__ bias, float* __restrict__ out) {
  const int lane   = threadIdx.x & 31;
  const int wid    = threadIdx.x >> 5;
  const int tile   = blockIdx.x * 4 + wid;        // 0..4095
  const int nTile  = tile % (CC / 64);            // 16 tiles of 64 cols
  const int mTile  = tile / (CC / 64);
  const int row0   = mTile * 16;
  const int col0   = nTile * 64;
  const int laneHi = lane >> 4;
  const int ln     = lane & 15;

  v8f acc[4] = {};
  const _Float16* arow = yh + (size_t)(row0 + ln) * CC;
  for (int k0 = 0; k0 < CC; k0 += 32) {
    const int ab = k0 + (laneHi ? 16 : 0);
    v8h alo = *(const v8h*)(arow + ab);
    v8h ahi = *(const v8h*)(arow + ab + 8);
    v16h afrag = CAT16(alo, ahi);
#pragma unroll
    for (int t = 0; t < 4; ++t) {
      const _Float16* bp = wpt + (size_t)(col0 + t * 16 + ln) * CC + k0 + (laneHi ? 16 : 0);
      v16h bfrag = *(const v16h*)bp;
      acc[t] = wmma_f16(afrag, bfrag, acc[t]);
    }
  }
#pragma unroll
  for (int t = 0; t < 4; ++t) {
    const int j    = col0 + t * 16 + ln;
    const float bj = bias[j];
#pragma unroll
    for (int r = 0; r < 8; ++r) {
      const int row = row0 + r + laneHi * 8;
      out[(size_t)row * CC + j] = acc[t][r] + bj;
    }
  }
}

// ---------------- launch ----------------

extern "C" void kernel_launch(void* const* d_in, const int* in_sizes, int n_in,
                              void* d_out, int out_size, void* d_ws, size_t ws_size,
                              hipStream_t stream) {
  (void)in_sizes; (void)n_in; (void)out_size; (void)ws_size;
  const float* x     = (const float*)d_in[0];   // [B,T,C]
  const float* Wkqv  = (const float*)d_in[1];   // [C,3C]
  const float* bkqv  = (const float*)d_in[2];   // [3C]
  const float* Wproj = (const float*)d_in[3];   // [C,C]
  const float* bproj = (const float*)d_in[4];   // [C]

  float* out  = (float*)d_out;                  // [B,T,C]
  float* kout = out + (size_t)MROWS * CC;       // [B,H,T,D]
  float* vout = kout + (size_t)MROWS * CC;      // [B,H,T,D]

  _Float16* wsh = (_Float16*)d_ws;
  _Float16* XH  = wsh;                          // 4,194,304 halves: x f16
  _Float16* WT  = wsh + 4194304;                // 3,145,728: W_kqv^T f16 [3C][C]
  _Float16* WPT = wsh + 7340032;                // 1,048,576: W_proj^T f16 [C][C]
  _Float16* QH  = wsh + 8388608;                // 4,194,304: q f16 [B,H,T,D]
  _Float16* KH  = wsh + 12582912;               // 4,194,304: k f16 [B,H,T,D]
  _Float16* VTH = wsh + 16777216;               // 4,194,304: v^T f16 [B,H,D,T]
  _Float16* YH  = wsh + 20971520;               // 4,194,304: attn out f16 [B,T,C]

  // 1. convert operands to f16 (weights transposed for contiguous B-fragments)
  f32_to_f16_kernel<<<(MROWS * CC) / 256, 256, 0, stream>>>(x, XH, MROWS * CC);
  transpose_f16_kernel<<<(CC * NKQV) / 256, 256, 0, stream>>>(Wkqv, WT, CC, NKQV);
  transpose_f16_kernel<<<(CC * CC) / 256, 256, 0, stream>>>(Wproj, WPT, CC, CC);

  // 2. QKV projection (writes k/v f32 outputs + f16 copies for attention)
  {
    const int tiles = (MROWS / 16) * (NKQV / 64);   // 12288
    qkv_gemm_kernel<<<tiles / 4, 128, 0, stream>>>(XH, WT, bkqv, QH, kout, KH, vout, VTH);
  }

  // 3. flash attention: workgroup = (b, h, 4 q-blocks), async-LDS staged K/V
  {
    const int groups = BB * HH * (TT / 16) / 4;     // 1024
    attn_kernel<<<groups, 128, 0, stream>>>(QH, KH, VTH, YH);
  }

  // 4. output projection
  {
    const int tiles = (MROWS / 16) * (CC / 64);     // 4096
    proj_gemm_kernel<<<tiles / 4, 128, 0, stream>>>(YH, WPT, bproj, out);
  }
}
